// HeteroGAT_80719615361488
// MI455X (gfx1250) — matrix operand order repaired and checked by reading, not compiled, not run on hardware
//
#include <hip/hip_runtime.h>
#include <hip/hip_bf16.h>

// ---------------- problem constants (match reference) ----------------
#define C_NU 100000
#define C_ND 150000
#define C_NI 120000
#define C_NP 80000
#define C_E  500000
#define HID  128

typedef __attribute__((ext_vector_type(16))) _Float16 v16h;
typedef __attribute__((ext_vector_type(8)))  float    v8f;

union Frag16 { v16h v; unsigned u[8]; };

#define LDT 34   // LDS row stride in halfs (even -> dword-aligned pair loads)

// =====================================================================
// Tiled WMMA GEMM:  C[N,128] = act(A[N,K] @ W[K,128] + bias)
// Optionally fuses attention dots: attn_out[n,h] += sum_d C_raw[n,h*32+d]*attn[h*32+d]
// Block = 256 threads (8 waves). Tile = 128 rows x 128 cols, K-step 32.
// =====================================================================
__global__ void __launch_bounds__(256)
hgat_wmma_gemm(const float* __restrict__ A, const float* __restrict__ W,
               const float* __restrict__ bias, float* __restrict__ C,
               const float* __restrict__ attn, float* __restrict__ attn_out,
               int N, int K, int actRelu, int writeC)
{
    __shared__ _Float16 ldsA[128 * LDT];   // A tile, row-major [row][k]
    __shared__ _Float16 ldsB[128 * LDT];   // W tile, transposed [col][k]

    const int tid  = threadIdx.x;
    const int lane = tid & 31;
    const int wave = tid >> 5;
    const int half = lane >> 4;
    const int l16  = lane & 15;
    const long rBase = (long)blockIdx.x * 128;

    v8f acc[8];
#pragma unroll
    for (int t = 0; t < 8; ++t) {
        v8f z = {0.f,0.f,0.f,0.f,0.f,0.f,0.f,0.f};
        acc[t] = z;
    }

    for (int k0 = 0; k0 < K; k0 += 32) {
        // ---- stage A tile (128x32) f32 -> f16, coalesced ----
#pragma unroll
        for (int i = 0; i < 16; ++i) {
            int idx = i * 256 + tid;            // 0..4095
            int r   = idx >> 5;
            int kk  = idx & 31;
            long row = rBase + r;
            float v = (row < N) ? A[row * (long)K + (k0 + kk)] : 0.f;
            ldsA[r * LDT + kk] = (_Float16)v;
        }
        // ---- stage W tile (32x128) transposed to [col][k] ----
#pragma unroll
        for (int i = 0; i < 16; ++i) {
            int idx = i * 256 + tid;
            int kk  = idx >> 7;                 // 0..31
            int col = idx & 127;
            float v = W[(long)(k0 + kk) * HID + col];
            ldsB[col * LDT + kk] = (_Float16)v;
        }
        __syncthreads();

        // ---- A fragment (16x32 f16, ISA 7.12.2 layout) ----
        Frag16 af;
        const unsigned* aU = (const unsigned*)(ldsA + (wave * 16 + l16) * LDT);
#pragma unroll
        for (int p = 0; p < 8; ++p) {
            int kk = ((p & 4) << 2) + (half << 3) + ((p & 3) << 1);
            af.u[p] = aU[kk >> 1];
        }
        // ---- 8 N-tiles of B, accumulate ----
#pragma unroll
        for (int t = 0; t < 8; ++t) {
            Frag16 bf;
            const unsigned* bU = (const unsigned*)(ldsB + (t * 16 + l16) * LDT);
#pragma unroll
            for (int p = 0; p < 8; ++p)
                bf.u[p] = bU[((p & 4) << 1) + (half << 2) + (p & 3)];   // table order: K=8h+s (s<8), 16+8h+(s-8) (s>=8)
            acc[t] = __builtin_amdgcn_wmma_f32_16x16x32_f16(
                false, af.v, false, bf.v, (short)0, acc[t], false, false);
        }
        __syncthreads();
    }

    const int mrow = wave * 16 + half * 8;   // + r gives the row inside tile

    if (writeC) {
#pragma unroll
        for (int t = 0; t < 8; ++t) {
            int col = t * 16 + l16;
            float bv = bias ? bias[col] : 0.f;
#pragma unroll
            for (int r = 0; r < 8; ++r) {
                long row = rBase + mrow + r;
                if (row < N) {
                    float v = acc[t][r] + bv;
                    if (actRelu) v = v > 0.f ? v : 0.f;
                    C[row * HID + col] = v;
                }
            }
        }
    }
    if (attn_out) {
        // el/er: per-row dot with attention vector; 16 partial lanes per (row,head)
        float av[8];
#pragma unroll
        for (int t = 0; t < 8; ++t) av[t] = attn[t * 16 + l16];
#pragma unroll
        for (int r = 0; r < 8; ++r) {
            long row = rBase + mrow + r;
            if (row < N) {
#pragma unroll
                for (int h = 0; h < 4; ++h) {
                    float pv = acc[2 * h][r] * av[2 * h] + acc[2 * h + 1][r] * av[2 * h + 1];
                    atomicAdd(&attn_out[row * 4 + h], pv);
                }
            }
        }
    }
}

// ------------------- monotone float<->uint key for atomic max -------------------
__device__ __forceinline__ unsigned f2key(float f) {
    unsigned b = __float_as_uint(f);
    return (b & 0x80000000u) ? ~b : (b | 0x80000000u);
}
__device__ __forceinline__ float key2f(unsigned k) {
    return (k & 0x80000000u) ? __uint_as_float(k & 0x7FFFFFFFu) : __uint_as_float(~k);
}
__device__ __forceinline__ float lrelu02(float x) { return x > 0.f ? x : 0.2f * x; }

// segment max over edges via integer atomicMax on ordered keys
__global__ void hgat_edge_max(const int* __restrict__ src, const int* __restrict__ dst,
                              const float* __restrict__ el, const float* __restrict__ er,
                              unsigned* __restrict__ MK, int E4)
{
    int i = blockIdx.x * blockDim.x + threadIdx.x;
    if (i >= E4) return;
    int e = i >> 2, h = i & 3;
    int s = src[e], d = dst[e];
    float v = lrelu02(el[s * 4 + h] + er[d * 4 + h]);
    atomicMax(&MK[d * 4 + h], f2key(v));
}

// key==0 -> empty segment -> 0.0 (matches isfinite guard in reference)
__global__ void hgat_decode_max(unsigned* __restrict__ MK, int n)
{
    int i = blockIdx.x * blockDim.x + threadIdx.x;
    if (i >= n) return;
    unsigned k = MK[i];
    ((float*)MK)[i] = (k == 0u) ? 0.f : key2f(k);
}

__global__ void hgat_edge_expsum(const int* __restrict__ src, const int* __restrict__ dst,
                                 const float* __restrict__ el, const float* __restrict__ er,
                                 const float* __restrict__ Mf, float* __restrict__ S,
                                 float* __restrict__ EX, int E4)
{
    int i = blockIdx.x * blockDim.x + threadIdx.x;
    if (i >= E4) return;
    int e = i >> 2, h = i & 3;
    int s = src[e], d = dst[e];
    float v  = lrelu02(el[s * 4 + h] + er[d * 4 + h]);
    float ex = __expf(v - Mf[d * 4 + h]);
    EX[i] = ex;
    atomicAdd(&S[d * 4 + h], ex);
}

// one wave (32 lanes) per edge; lane handles dims {lane, lane+32, lane+64, lane+96}
__global__ void hgat_edge_agg(const int* __restrict__ src, const int* __restrict__ dst,
                              const float* __restrict__ FS, const float* __restrict__ EX,
                              const float* __restrict__ S, float* __restrict__ RST, int E)
{
    int gid  = blockIdx.x * blockDim.x + threadIdx.x;
    int e    = gid >> 5;
    int lane = gid & 31;
    if (e >= E) return;
    int s = src[e], d = dst[e];
    const float* fsr = FS + (size_t)s * HID;
    float* outr = RST + (size_t)d * HID;
#pragma unroll
    for (int j = 0; j < 4; ++j) {                 // j == head index
        float a = EX[(size_t)e * 4 + j] / S[(size_t)d * 4 + j];
        int dd = j * 32 + lane;
        atomicAdd(&outr[dd], fsr[dd] * a);
    }
}

__global__ void hgat_bias_elu(float* __restrict__ H, const float* __restrict__ bc, int n)
{
    int i = blockIdx.x * blockDim.x + threadIdx.x;
    if (i >= n) return;
    float v = H[i] + bc[i & (HID - 1)];
    H[i] = v > 0.f ? v : (__expf(v) - 1.f);
}

__global__ void hgat_head2(const float* __restrict__ T, const float* __restrict__ Wh2,
                           const float* __restrict__ bh2, float* __restrict__ out, int n2)
{
    int i = blockIdx.x * blockDim.x + threadIdx.x;
    if (i >= n2) return;
    int n = i >> 1, c = i & 1;
    float s = bh2[c];
    const float* t = T + (size_t)n * HID;
#pragma unroll 8
    for (int k = 0; k < HID; ++k) s += t[k] * Wh2[k * 2 + c];
    out[i] = s;
}

// =====================================================================
extern "C" void kernel_launch(void* const* d_in, const int* in_sizes, int n_in,
                              void* d_out, int out_size, void* d_ws, size_t ws_size,
                              hipStream_t stream)
{
    (void)in_sizes; (void)n_in; (void)out_size; (void)ws_size;

    const float* x_user   = (const float*)d_in[0];
    const float* x_device = (const float*)d_in[1];
    const float* x_ip     = (const float*)d_in[2];
    const float* x_phone  = (const float*)d_in[3];
    const int* src_ud = (const int*)d_in[4];  const int* dst_ud = (const int*)d_in[5];
    const int* src_du = (const int*)d_in[6];  const int* dst_du = (const int*)d_in[7];
    const int* src_ui = (const int*)d_in[8];  const int* dst_ui = (const int*)d_in[9];
    const int* src_up = (const int*)d_in[10]; const int* dst_up = (const int*)d_in[11];
    const float* Wp_user   = (const float*)d_in[12]; const float* bp_user   = (const float*)d_in[13];
    const float* Wp_device = (const float*)d_in[14]; const float* bp_device = (const float*)d_in[15];
    const float* Wp_ip     = (const float*)d_in[16]; const float* bp_ip     = (const float*)d_in[17];
    const float* Wp_phone  = (const float*)d_in[18]; const float* bp_phone  = (const float*)d_in[19];
    const float* Wc1 = (const float*)d_in[20]; const float* al1 = (const float*)d_in[21];
    const float* ar1 = (const float*)d_in[22]; const float* bc1 = (const float*)d_in[23];
    const float* Wc2 = (const float*)d_in[24]; const float* al2 = (const float*)d_in[25];
    const float* ar2 = (const float*)d_in[26]; const float* bc2 = (const float*)d_in[27];
    const float* Wh1 = (const float*)d_in[28]; const float* bh1 = (const float*)d_in[29];
    const float* Wh2 = (const float*)d_in[30]; const float* bh2 = (const float*)d_in[31];

    // ---------------- workspace carve-up (floats) ----------------
    float* ws = (float*)d_ws;
    size_t off = 0;
    auto carve = [&](size_t n) { float* p = ws + off; off += n; return p; };
    float* H0u = carve((size_t)C_NU * HID);
    float* H0d = carve((size_t)C_ND * HID);
    float* H0i = carve((size_t)C_NI * HID);
    float* H0p = carve((size_t)C_NP * HID);
    float* H1u = carve((size_t)C_NU * HID);
    float* H1d = carve((size_t)C_ND * HID);
    float* H1i = carve((size_t)C_NI * HID);
    float* H1p = carve((size_t)C_NP * HID);
    float* FS  = carve((size_t)C_ND * HID);      // max Ns = 150000
    float* EL  = carve((size_t)C_ND * 4);
    float* ER  = carve((size_t)C_ND * 4);
    float* MKf = carve((size_t)C_ND * 4);        // uint keys then decoded floats
    float* Sg  = carve((size_t)C_ND * 4);
    float* EX  = carve((size_t)C_E * 4);

    const size_t hTotal = ((size_t)C_NU + C_ND + C_NI + C_NP) * HID; // one ping-pong set

    auto gemm = [&](const float* A, const float* W, const float* bias, float* C,
                    const float* attn, float* attn_out, int N, int K, int act, int writeC) {
        int grid = (N + 127) / 128;
        hgat_wmma_gemm<<<grid, 256, 0, stream>>>(A, W, bias, C, attn, attn_out,
                                                 N, K, act, writeC);
    };

    // ---------------- input projections ----------------
    gemm(x_user,   Wp_user,   bp_user,   H0u, nullptr, nullptr, C_NU, 128, 0, 1);
    gemm(x_device, Wp_device, bp_device, H0d, nullptr, nullptr, C_ND, 64,  0, 1);
    gemm(x_ip,     Wp_ip,     bp_ip,     H0i, nullptr, nullptr, C_NI, 64,  0, 1);
    gemm(x_phone,  Wp_phone,  bp_phone,  H0p, nullptr, nullptr, C_NP, 64,  0, 1);

    // ---------------- one GAT relation ----------------
    auto relation = [&](const float* hsrc, int Ns, const float* hdst, int Nd,
                        const int* se, const int* de, const float* Wc_r,
                        const float* al_r, const float* ar_r, const float* bc_r,
                        float* outDst) {
        hipMemsetAsync(EL,  0, (size_t)Ns * 4 * sizeof(float), stream);
        hipMemsetAsync(ER,  0, (size_t)Nd * 4 * sizeof(float), stream);
        hipMemsetAsync(MKf, 0, (size_t)Nd * 4 * sizeof(float), stream);
        hipMemsetAsync(Sg,  0, (size_t)Nd * 4 * sizeof(float), stream);

        // fs = hsrc @ W (kept), el fused; fd @ W not stored, only er fused
        gemm(hsrc, Wc_r, nullptr, FS,      al_r, EL, Ns, 128, 0, 1);
        gemm(hdst, Wc_r, nullptr, nullptr, ar_r, ER, Nd, 128, 0, 0);

        int e4 = C_E * 4;
        int gb = (e4 + 255) / 256;
        hgat_edge_max<<<gb, 256, 0, stream>>>(se, de, EL, ER, (unsigned*)MKf, e4);
        hgat_decode_max<<<(Nd * 4 + 255) / 256, 256, 0, stream>>>((unsigned*)MKf, Nd * 4);
        hgat_edge_expsum<<<gb, 256, 0, stream>>>(se, de, EL, ER, MKf, Sg, EX, e4);
        hgat_edge_agg<<<(C_E * 32 + 255) / 256, 256, 0, stream>>>(se, de, FS, EX, Sg, outDst, C_E);
        hgat_bias_elu<<<(Nd * HID + 255) / 256, 256, 0, stream>>>(outDst, bc_r, Nd * HID);
    };

    auto conv = [&](const float* Wc, const float* al, const float* ar, const float* bc,
                    const float* hu, const float* hd, const float* hi, const float* hp,
                    float* nu, float* nd_, float* ni, float* np_) {
        // zero all 4 dst accumulators (contiguous set)
        hipMemsetAsync(nu, 0, hTotal * sizeof(float), stream);
        relation(hu, C_NU, hd, C_ND, src_ud, dst_ud, Wc + 0 * 128 * 128, al + 0 * 128, ar + 0 * 128, bc + 0 * 128, nd_);
        relation(hd, C_ND, hu, C_NU, src_du, dst_du, Wc + 1 * 128 * 128, al + 1 * 128, ar + 1 * 128, bc + 1 * 128, nu);
        relation(hu, C_NU, hi, C_NI, src_ui, dst_ui, Wc + 2 * 128 * 128, al + 2 * 128, ar + 2 * 128, bc + 2 * 128, ni);
        relation(hu, C_NU, hp, C_NP, src_up, dst_up, Wc + 3 * 128 * 128, al + 3 * 128, ar + 3 * 128, bc + 3 * 128, np_);
    };

    // layer 1: H0 -> H1 ; layer 2: H1 -> H0
    conv(Wc1, al1, ar1, bc1, H0u, H0d, H0i, H0p, H1u, H1d, H1i, H1p);
    conv(Wc2, al2, ar2, bc2, H1u, H1d, H1i, H1p, H0u, H0d, H0i, H0p);

    // ---------------- user head MLP ----------------
    gemm(H0u, Wh1, bh1, FS, nullptr, nullptr, C_NU, 128, 1, 1);   // relu(hu@Wh1+bh1) -> FS
    hgat_head2<<<(C_NU * 2 + 255) / 256, 256, 0, stream>>>(FS, Wh2, bh2, (float*)d_out, C_NU * 2);
}
